// HarmonicLoss_62113817035273
// MI455X (gfx1250) — compile-verified, matrix operand order checked by reading, not run on hardware
//
#include <hip/hip_runtime.h>
#include <stdint.h>

// ---------------- problem constants ----------------
#define B_ROWS   4096
#define C_COLS   32000
#define D_DIM    1024
#define N_EXP    10.0f
#define EPS_F    1e-8f
#define NORM_EPS 1e-12f

// ---------------- tiling ----------------
#define MT 64            // rows per block tile
#define NT 256           // cols per block tile
#define KT 32            // K per WMMA step (bf16)
#define K_ITERS (D_DIM / KT)    // 32
#define N_TILES (C_COLS / NT)   // 125
#define M_TILES (B_ROWS / MT)   // 64
#define LDSK 40          // padded K stride in ushorts (80B, 16B-aligned rows)

typedef __attribute__((ext_vector_type(8)))  float  v8f;
typedef __attribute__((ext_vector_type(16))) __bf16 v16bf;
typedef int v4i_ __attribute__((vector_size(16)));   // matches builtin's b128 pointee

union Frag16 { v16bf v; uint32_t u[8]; };

#if __has_builtin(__builtin_amdgcn_global_load_async_to_lds_b128)
#define HAS_ASYNC_LDS 1
#else
#define HAS_ASYNC_LDS 0
#endif

static __device__ __forceinline__ unsigned short f32_to_bf16(float f) {
  uint32_t u = __float_as_uint(f);
  uint32_t r = 0x7FFFu + ((u >> 16) & 1u);   // round-to-nearest-even
  return (unsigned short)((u + r) >> 16);
}
static __device__ __forceinline__ float bf16_to_f32(unsigned short h) {
  return __uint_as_float(((uint32_t)h) << 16);
}
static __device__ __forceinline__ float harmonic_logit(float sim) {
  float dsq = fmaxf(2.0f - 2.0f * sim, 0.0f);
  return -N_EXP * logf(sqrtf(dsq) + EPS_F);
}

#if HAS_ASYNC_LDS
static __device__ __forceinline__ void async_b128(const unsigned short* g,
                                                  unsigned short* l) {
  __builtin_amdgcn_global_load_async_to_lds_b128(
      (__attribute__((address_space(1))) v4i_*)g,
      (__attribute__((address_space(3))) v4i_*)l, 0, 0);
}
static __device__ __forceinline__ void wait_async_le5() {
#if __has_builtin(__builtin_amdgcn_s_wait_asynccnt)
  __builtin_amdgcn_s_wait_asynccnt(5);
#else
  asm volatile("s_wait_asynccnt 0x5" ::: "memory");
#endif
}
static __device__ __forceinline__ void wait_async_le0() {
#if __has_builtin(__builtin_amdgcn_s_wait_asynccnt)
  __builtin_amdgcn_s_wait_asynccnt(0);
#else
  asm volatile("s_wait_asynccnt 0x0" ::: "memory");
#endif
}
#endif

// ---------------- kernel 1: L2-normalize rows, emit bf16 ----------------
__global__ __launch_bounds__(256)
void normalize_rows(const float* __restrict__ src, unsigned short* __restrict__ dst) {
  const int row = blockIdx.x;
  const int t = threadIdx.x;
  const float4 v = ((const float4*)(src + (size_t)row * D_DIM))[t];
  float ss = v.x * v.x + v.y * v.y + v.z * v.z + v.w * v.w;
  #pragma unroll
  for (int m = 16; m >= 1; m >>= 1) ss += __shfl_xor(ss, m, 32);
  __shared__ float wss[8];
  if ((t & 31) == 0) wss[t >> 5] = ss;
  __syncthreads();
  float tot = 0.0f;
  #pragma unroll
  for (int i = 0; i < 8; ++i) tot += wss[i];
  const float inv = 1.0f / fmaxf(sqrtf(tot), NORM_EPS);
  uint2 o;
  o.x = (uint32_t)f32_to_bf16(v.x * inv) | ((uint32_t)f32_to_bf16(v.y * inv) << 16);
  o.y = (uint32_t)f32_to_bf16(v.z * inv) | ((uint32_t)f32_to_bf16(v.w * inv) << 16);
  ((uint2*)(dst + (size_t)row * D_DIM))[t] = o;
}

// ---------------- kernel 2: fused bf16 WMMA GEMM + per-tile online LSE ----------------
// grid = (N_TILES, M_TILES), block = 256 (8 waves). Each wave: 2(M) x 4(N) 16x16 frags.
// Triple-buffered LDS, depth-2 async global->LDS pipeline, 1 barrier/iter.
__global__ __launch_bounds__(256)
void fused_gemm_lse(const unsigned short* __restrict__ xn,
                    const unsigned short* __restrict__ wn,
                    float* __restrict__ pmax, float* __restrict__ psum) {
  __shared__ __align__(16) unsigned short As[3][MT][LDSK];
  __shared__ __align__(16) unsigned short Bs[3][NT][LDSK];
  __shared__ float redm[4][MT];
  __shared__ float reds[4][MT];

  const int t      = threadIdx.x;
  const int lane   = t & 31;
  const int wid    = t >> 5;
  const int wave_m = wid >> 2;        // 0..1 -> 32-row slab
  const int wave_n = wid & 3;         // 0..3 -> 64-col slab
  const int lhalf  = lane >> 4;       // 0/1
  const int l15    = lane & 15;
  const int khalf  = lhalf * 8;       // ISA 16-bit A/B lane->K split

  const int m0 = blockIdx.y * MT;
  const int n0 = blockIdx.x * NT;

  v8f acc[2][4] = {};

  const int am = t >> 2;              // 0..63: A-tile row this thread stages
  const int ak = (t & 3) * 8;         // 0/8/16/24: 8 bf16 per 16B transfer

  const unsigned short* ga = xn + (size_t)(m0 + am) * D_DIM + ak;  // + kk
  const unsigned short* gb = wn + (size_t)(n0 + t) * D_DIM;        // + kk + j*8

#if HAS_ASYNC_LDS
  // ---- async staging helper: 1 A-chunk + 4 B-chunks of 16B per thread ----
  #define STAGE(buf, kk)                                                  \
    do {                                                                  \
      async_b128(ga + (kk), &As[(buf)][am][ak]);                          \
      _Pragma("unroll")                                                   \
      for (int j = 0; j < 4; ++j)                                         \
        async_b128(gb + (kk) + j * 8, &Bs[(buf)][t][j * 8]);              \
    } while (0)

  STAGE(0, 0);
  STAGE(1, KT);

  int cur = 0, nxt = 1, fut = 2;
  for (int it = 0; it < K_ITERS; ++it) {
    if (it < K_ITERS - 1) wait_async_le5();   // own loads for `cur` complete
    else                  wait_async_le0();
    __syncthreads();                          // everyone's `cur` data in LDS
    if (it + 2 < K_ITERS) STAGE(fut, (it + 2) * KT);
#else
  int cur = 0;
  for (int it = 0; it < K_ITERS; ++it) {
    const int kk = it * KT;
    const uint4 a_g = *(const uint4*)(ga + kk);
    uint4 b_g[4];
    #pragma unroll
    for (int j = 0; j < 4; ++j) b_g[j] = *(const uint4*)(gb + kk + j * 8);
    __syncthreads();
    *(uint4*)&As[0][am][ak] = a_g;
    #pragma unroll
    for (int j = 0; j < 4; ++j) *(uint4*)&Bs[0][t][j * 8] = b_g[j];
    __syncthreads();
#endif

    // build frags per ISA 16-bit layout: VGPR0-3 <- K[khalf..+7], VGPR4-7 <- K[16+khalf..]
    Frag16 fa[2], fb[4];
    #pragma unroll
    for (int mf = 0; mf < 2; ++mf) {
      const unsigned short* p = &As[cur][wave_m * 32 + mf * 16 + l15][0];
      const uint32_t* q0 = (const uint32_t*)(p + khalf);
      const uint32_t* q1 = (const uint32_t*)(p + 16 + khalf);
      #pragma unroll
      for (int i = 0; i < 4; ++i) { fa[mf].u[i] = q0[i]; fa[mf].u[4 + i] = q1[i]; }
    }
    #pragma unroll
    for (int nf = 0; nf < 4; ++nf) {
      const unsigned short* p = &Bs[cur][wave_n * 64 + nf * 16 + l15][0];
      const uint32_t* q0 = (const uint32_t*)(p + khalf);
      const uint32_t* q1 = (const uint32_t*)(p + 16 + khalf);
      #pragma unroll
      for (int i = 0; i < 4; ++i) { fb[nf].u[i] = q0[i]; fb[nf].u[4 + i] = q1[i]; }
    }

    #pragma unroll
    for (int mf = 0; mf < 2; ++mf)
      #pragma unroll
      for (int nf = 0; nf < 4; ++nf)
        acc[mf][nf] = __builtin_amdgcn_wmma_f32_16x16x32_bf16(
            false, fa[mf].v, false, fb[nf].v, (short)0, acc[mf][nf], false, false);

#if HAS_ASYNC_LDS
    const int tmp = cur; cur = nxt; nxt = fut; fut = tmp;   // rotate buffers
#endif
  }

  // ---- epilogue: sim -> logit, row-wise (max, sumexp) over this 256-col tile ----
  #pragma unroll
  for (int mf = 0; mf < 2; ++mf) {
    #pragma unroll
    for (int v = 0; v < 8; ++v) {
      float lv[4];
      float rmax = -__builtin_inff();
      #pragma unroll
      for (int nf = 0; nf < 4; ++nf) {
        lv[nf] = harmonic_logit(acc[mf][nf][v]);
        rmax = fmaxf(rmax, lv[nf]);
      }
      #pragma unroll
      for (int m = 1; m < 16; m <<= 1) rmax = fmaxf(rmax, __shfl_xor(rmax, m, 32));
      float se = 0.0f;
      #pragma unroll
      for (int nf = 0; nf < 4; ++nf) se += expf(lv[nf] - rmax);
      #pragma unroll
      for (int m = 1; m < 16; m <<= 1) se += __shfl_xor(se, m, 32);
      if (l15 == 0) {
        const int rrow = wave_m * 32 + mf * 16 + v + lhalf * 8;
        redm[wave_n][rrow] = rmax;
        reds[wave_n][rrow] = se;
      }
    }
  }
  __syncthreads();
  if (t < MT) {
    float M = redm[0][t];
    #pragma unroll
    for (int i = 1; i < 4; ++i) M = fmaxf(M, redm[i][t]);
    float S = 0.0f;
    #pragma unroll
    for (int i = 0; i < 4; ++i) S += reds[i][t] * expf(redm[i][t] - M);
    const size_t gi = (size_t)(m0 + t) * N_TILES + blockIdx.x;
    pmax[gi] = M;
    psum[gi] = S;
  }
}

// ---------------- kernel 3: logit at target column (one wave per row) ----------------
__global__ __launch_bounds__(256)
void target_logits(const unsigned short* __restrict__ xn,
                   const unsigned short* __restrict__ wn,
                   const int* __restrict__ targets, float* __restrict__ tlogit) {
  const int lane = threadIdx.x & 31;
  const int b = blockIdx.x * 8 + (threadIdx.x >> 5);
  const int tgt = targets[b];
  const unsigned short* xr = xn + (size_t)b * D_DIM;
  const unsigned short* wr = wn + (size_t)tgt * D_DIM;
  float dot = 0.0f;
  for (int k = lane * 4; k < D_DIM; k += 32 * 4) {
    const uint2 xa = *(const uint2*)(xr + k);
    const uint2 wa = *(const uint2*)(wr + k);
    dot += bf16_to_f32((unsigned short)(xa.x & 0xFFFF)) * bf16_to_f32((unsigned short)(wa.x & 0xFFFF));
    dot += bf16_to_f32((unsigned short)(xa.x >> 16))    * bf16_to_f32((unsigned short)(wa.x >> 16));
    dot += bf16_to_f32((unsigned short)(xa.y & 0xFFFF)) * bf16_to_f32((unsigned short)(wa.y & 0xFFFF));
    dot += bf16_to_f32((unsigned short)(xa.y >> 16))    * bf16_to_f32((unsigned short)(wa.y >> 16));
  }
  #pragma unroll
  for (int m = 16; m >= 1; m >>= 1) dot += __shfl_xor(dot, m, 32);
  if (lane == 0) tlogit[b] = harmonic_logit(dot);
}

// ---------------- kernel 4: merge per-tile partials into per-row loss ----------------
__global__ __launch_bounds__(256)
void row_merge(const float* __restrict__ pmax, const float* __restrict__ psum,
               const float* __restrict__ tlogit, float* __restrict__ rloss) {
  const int b = blockIdx.x * blockDim.x + threadIdx.x;
  if (b >= B_ROWS) return;
  const float* pm = pmax + (size_t)b * N_TILES;
  const float* ps = psum + (size_t)b * N_TILES;
  float M = -__builtin_inff();
  for (int i = 0; i < N_TILES; ++i) M = fmaxf(M, pm[i]);
  float S = 0.0f;
  for (int i = 0; i < N_TILES; ++i) S += ps[i] * expf(pm[i] - M);
  const float lse = M + logf(S);
  const float p = expf(tlogit[b] - lse);
  rloss[b] = -logf(p + EPS_F);
}

// ---------------- kernel 5: deterministic mean ----------------
__global__ __launch_bounds__(256)
void final_reduce(const float* __restrict__ rloss, float* __restrict__ out) {
  __shared__ float sm[256];
  const int t = threadIdx.x;
  float s = 0.0f;
  for (int i = t; i < B_ROWS; i += 256) s += rloss[i];
  sm[t] = s;
  __syncthreads();
  for (int o = 128; o > 0; o >>= 1) {
    if (t < o) sm[t] += sm[t + o];
    __syncthreads();
  }
  if (t == 0) out[0] = sm[0] / (float)B_ROWS;
}

// ---------------- launch ----------------
extern "C" void kernel_launch(void* const* d_in, const int* in_sizes, int n_in,
                              void* d_out, int out_size, void* d_ws, size_t ws_size,
                              hipStream_t stream) {
  const float* x = (const float*)d_in[0];
  const float* w = (const float*)d_in[1];
  const int* targets = (const int*)d_in[2];
  float* out = (float*)d_out;

  char* ws = (char*)d_ws;
  const size_t XN_BYTES = (size_t)B_ROWS * D_DIM * 2;          //  8 MB
  const size_t WN_BYTES = (size_t)C_COLS * D_DIM * 2;          // 64 MB
  const size_t P_BYTES  = (size_t)B_ROWS * N_TILES * 4;        //  2 MB each
  unsigned short* xn = (unsigned short*)ws;
  unsigned short* wn = (unsigned short*)(ws + XN_BYTES);
  float* pmax   = (float*)(ws + XN_BYTES + WN_BYTES);
  float* psum   = (float*)(ws + XN_BYTES + WN_BYTES + P_BYTES);
  float* tlogit = (float*)(ws + XN_BYTES + WN_BYTES + 2 * P_BYTES);
  float* rloss  = (float*)(ws + XN_BYTES + WN_BYTES + 2 * P_BYTES + (size_t)B_ROWS * 4);

  normalize_rows<<<B_ROWS, 256, 0, stream>>>(x, xn);
  normalize_rows<<<C_COLS, 256, 0, stream>>>(w, wn);

  dim3 grid(N_TILES, M_TILES);
  fused_gemm_lse<<<grid, 256, 0, stream>>>(xn, wn, pmax, psum);

  target_logits<<<B_ROWS / 8, 256, 0, stream>>>(xn, wn, targets, tlogit);
  row_merge<<<(B_ROWS + 255) / 256, 256, 0, stream>>>(pmax, psum, tlogit, rloss);
  final_reduce<<<1, 256, 0, stream>>>(rloss, out);
}
